// SpatialAttention_69956427317927
// MI455X (gfx1250) — matrix operand order repaired and checked by reading, not compiled, not run on hardware
//
#include <hip/hip_runtime.h>
#include <hip/hip_bf16.h>

typedef __attribute__((ext_vector_type(16))) _Float16 v16h;
typedef __attribute__((ext_vector_type(8)))  float    v8f;

#define BT     16      // B*T
#define SEQ    1024    // N
#define DMODEL 512     // D
#define NH     8       // heads
#define HD     64      // head dim
#define D3     1536    // 3*D

static __device__ inline v8f wmma16(v16h a, v16h b, v8f c) {
  // v_wmma_f32_16x16x32_f16: D = A(16x32 f16) * B(32x16 f16) + C(16x16 f32)
  return __builtin_amdgcn_wmma_f32_16x16x32_f16(false, a, false, b, (short)0, c,
                                                false, false);
}

// A fragment: lane L holds row M = L%16; a[0..7] = K[kbase..kbase+7],
// a[8..15] = K[16+kbase..16+kbase+7], kbase = (L<16)?0:8.
static __device__ inline v16h load_a(const _Float16* __restrict__ row, int kbase) {
  v16h a;
#pragma unroll
  for (int i = 0; i < 8; ++i) {
    a[i]     = row[kbase + i];
    a[8 + i] = row[16 + kbase + i];
  }
  return a;
}

// B fragment: lane L holds col N = L%16; b[0..7] = B[kb/2+i][N], b[8..15] = B[16+kb/2+i][N], kb = (L<16)?0:16.
// `col` points at 32 contiguous f16 (tiles staged [n][k]).
static __device__ inline v16h load_b(const _Float16* __restrict__ col, int kb) {
  v16h b;
#pragma unroll
  for (int i = 0; i < 8; ++i) {
    b[i]     = col[(kb >> 1) + i];
    b[8 + i] = col[16 + (kb >> 1) + i];
  }
  return b;
}

// CDNA5 async DMA: copy 16 bytes from global to LDS, tracked by ASYNCcnt.
// lds_off is the wave-relative LDS byte offset (= low 32 bits of flat addr).
static __device__ inline void async_copy16(const _Float16* g, unsigned lds_off) {
  asm volatile("global_load_async_to_lds_b128 %0, %1, off"
               :: "v"(lds_off), "v"(g) : "memory");
}
static __device__ inline void wait_async0() {
  asm volatile("s_wait_asynccnt 0x0" ::: "memory");
}

// ---------------------------------------------------------------------------
// Kernel 1: qkv = x @ W_qkv + b_qkv.
// Q,K scattered as [bt][h][n][hd] f16; V scattered TRANSPOSED as [bt][h][e][n]
// so the attention kernel can stage both K and V^T with straight async DMA.
// Block = 128 threads (4 waves), tile 128(M) x 64(N); wave owns 32x64.
// ---------------------------------------------------------------------------
__global__ __launch_bounds__(128) void qkv_kernel(
    const float* __restrict__ X, const float* __restrict__ Wq,
    const float* __restrict__ bq,
    _Float16* __restrict__ Qo, _Float16* __restrict__ Ko,
    _Float16* __restrict__ Vo) {
  __shared__ _Float16 As[128][32];  // [m][k]
  __shared__ _Float16 Bs[64][32];   // [n][k] (W transposed on write)
  const int t = threadIdx.x;
  const int w = t >> 5, lane = t & 31;
  const int m0 = blockIdx.y * 128;
  const int n0 = blockIdx.x * 64;
  const int lr = lane & 15;
  const int kbase = (lane < 16) ? 0 : 8;
  const int kb    = (lane < 16) ? 0 : 16;
  const int half8 = (lane < 16) ? 0 : 8;

  v8f acc0[4] = {};
  v8f acc1[4] = {};

  for (int k0 = 0; k0 < DMODEL; k0 += 32) {
    __syncthreads();
#pragma unroll
    for (int i = 0; i < 32; ++i) {
      int idx = i * 128 + t;
      int r = idx >> 5, c = idx & 31;
      As[r][c] = (_Float16)X[(size_t)(m0 + r) * DMODEL + (k0 + c)];
    }
#pragma unroll
    for (int i = 0; i < 16; ++i) {
      int idx = i * 128 + t;
      int nn = idx & 63, kk = idx >> 6;
      Bs[nn][kk] = (_Float16)Wq[(size_t)(k0 + kk) * D3 + (n0 + nn)];
    }
    __syncthreads();
    v16h a0 = load_a(&As[w * 32 + lr][0], kbase);
    v16h a1 = load_a(&As[w * 32 + 16 + lr][0], kbase);
#pragma unroll
    for (int nt = 0; nt < 4; ++nt) {
      v16h b = load_b(&Bs[nt * 16 + lr][0], kb);
      acc0[nt] = wmma16(a0, b, acc0[nt]);
      acc1[nt] = wmma16(a1, b, acc1[nt]);
    }
  }

#pragma unroll
  for (int nt = 0; nt < 4; ++nt) {
    int col   = n0 + nt * 16 + lr;        // 0..1535
    int which = col / DMODEL;             // 0=q 1=k 2=v (uniform per 64-block)
    int h     = (col % DMODEL) / HD;
    int e     = col % HD;
    float bias = bq[col];
#pragma unroll
    for (int sub = 0; sub < 2; ++sub) {
      const v8f* acc = sub ? acc1 : acc0;
#pragma unroll
      for (int r = 0; r < 8; ++r) {
        int row = m0 + w * 32 + sub * 16 + r + half8;  // 0..16383
        int btf = row >> 10;
        int n   = row & (SEQ - 1);
        _Float16 val = (_Float16)(acc[nt][r] + bias);
        if (which == 2)       // V transposed: [bt][h][e][n]
          Vo[(((size_t)btf * NH + h) * HD + e) * SEQ + n] = val;
        else if (which == 1)  // K: [bt][h][n][hd]
          Ko[(((size_t)btf * NH + h) * SEQ + n) * HD + e] = val;
        else                  // Q: [bt][h][n][hd]
          Qo[(((size_t)btf * NH + h) * SEQ + n) * HD + e] = val;
      }
    }
  }
}

// ---------------------------------------------------------------------------
// Kernel 2: flash attention, one (bt,h,64-query) block; wave = 16 query rows.
// K and V^T chunks staged by double-buffered async global->LDS DMA.
// ---------------------------------------------------------------------------
__global__ __launch_bounds__(128) void attn_kernel(
    const _Float16* __restrict__ Qg, const _Float16* __restrict__ Kg,
    const _Float16* __restrict__ Vg, _Float16* __restrict__ AO) {
  __shared__ _Float16 Ks[2][32][64];    // [buf][key][hd]
  __shared__ _Float16 Vt[2][64][32];    // [buf][hd][key] (V stored transposed)
  __shared__ _Float16 Ps[4][16][32];    // per-wave P tile [qrow][key]
  const int t = threadIdx.x;
  const int w = t >> 5, lane = t & 31;
  const int bh   = blockIdx.x >> 4;     // bt*NH + h (0..127)
  const int qblk = blockIdx.x & 15;
  const int bt = bh >> 3, h = bh & 7;
  const size_t base = (size_t)bh * SEQ * HD;
  const _Float16* Qm  = Qg + base;
  const _Float16* Km  = Kg + base;      // [n][hd]
  const _Float16* Vtg = Vg + base;      // [e][n]
  const int lr = lane & 15;
  const int kbase = (lane < 16) ? 0 : 8;
  const int kb    = (lane < 16) ? 0 : 16;
  const int half8 = (lane < 16) ? 0 : 8;
  const int q0 = qblk * 64 + w * 16;

  // Q fragments over hd (two K=32 chunks), pre-scaled by hd^-0.5 = 1/8 (exact)
  v16h qa[2];
#pragma unroll
  for (int c = 0; c < 2; ++c) {
    const _Float16* qrow = Qm + (size_t)(q0 + lr) * HD + c * 32;
#pragma unroll
    for (int i = 0; i < 8; ++i) {
      qa[c][i]     = (_Float16)((float)qrow[kbase + i]      * 0.125f);
      qa[c][8 + i] = (_Float16)((float)qrow[16 + kbase + i] * 0.125f);
    }
  }

  v8f o[4] = {};
  float mrow[8], lrow[8];
#pragma unroll
  for (int r = 0; r < 8; ++r) { mrow[r] = -3.0e38f; lrow[r] = 0.0f; }

  // async-stage one 32-key chunk (4KB K + 4KB V^T) into buffer chunk&1
  auto issue = [&](int chunk) {
    const int bufi = chunk & 1;
    const int key0 = chunk * 32;
    const _Float16* ksrc = Km + (size_t)key0 * HD;   // 4KB contiguous
    const _Float16* vsrc = Vtg + key0;               // 64 rows x 64B, stride SEQ
    unsigned koff = (unsigned)(uintptr_t)&Ks[bufi][0][0];
    unsigned voff = (unsigned)(uintptr_t)&Vt[bufi][0][0];
#pragma unroll
    for (int j = 0; j < 2; ++j) {
      int idx = j * 128 + t;                          // 256 x 16B chunks
      async_copy16(ksrc + idx * 8, koff + (unsigned)idx * 16u);
      int e = idx >> 2, ko = (idx & 3) * 8;           // row e, key offset ko
      async_copy16(vsrc + (size_t)e * SEQ + ko,
                   voff + (unsigned)(e * 64 + ko * 2));
    }
  };

  issue(0);

  for (int i = 0; i < 32; ++i) {
    wait_async0();        // this wave's chunk-i DMA complete
    __syncthreads();      // all waves' DMA done; prev compute done everywhere
    if (i + 1 < 32) issue(i + 1);   // overlaps with compute below
    const int cur = i & 1;

    // S = (Q/8) @ K^T : rows of Ks are exactly B-operand columns of K^T
    v8f s0 = {}, s1 = {};
    {
      v16h b;
      b = load_b(&Ks[cur][lr][0],       kb); s0 = wmma16(qa[0], b, s0);
      b = load_b(&Ks[cur][lr][32],      kb); s0 = wmma16(qa[1], b, s0);
      b = load_b(&Ks[cur][16 + lr][0],  kb); s1 = wmma16(qa[0], b, s1);
      b = load_b(&Ks[cur][16 + lr][32], kb); s1 = wmma16(qa[1], b, s1);
    }

    // online softmax: C-layout row M=r lives in lanes 0-15, M=r+8 in 16-31,
    // so 16-lane shfl_xor reductions give the right per-row broadcast.
#pragma unroll
    for (int r = 0; r < 8; ++r) {
      float mx = fmaxf(s0[r], s1[r]);
#pragma unroll
      for (int msk = 8; msk >= 1; msk >>= 1)
        mx = fmaxf(mx, __shfl_xor(mx, msk, 32));
      float mn    = fmaxf(mrow[r], mx);
      float alpha = __expf(mrow[r] - mn);
      float p0 = __expf(s0[r] - mn);
      float p1 = __expf(s1[r] - mn);
      float rs = p0 + p1;
#pragma unroll
      for (int msk = 8; msk >= 1; msk >>= 1) rs += __shfl_xor(rs, msk, 32);
      lrow[r] = lrow[r] * alpha + rs;
      mrow[r] = mn;
#pragma unroll
      for (int nt = 0; nt < 4; ++nt) o[nt][r] *= alpha;
      int prow = r + half8;
      Ps[w][prow][lr]      = (_Float16)p0;
      Ps[w][prow][16 + lr] = (_Float16)p1;
    }
    // wave-private tile: DS ops are in-order per wave; wait + reorder fence
    asm volatile("s_wait_dscnt 0x0" ::: "memory");

    v16h pa = load_a(&Ps[w][lr][0], kbase);
#pragma unroll
    for (int nt = 0; nt < 4; ++nt) {
      v16h b = load_b(&Vt[cur][nt * 16 + lr][0], kb);
      o[nt] = wmma16(pa, b, o[nt]);
    }
  }

  // normalize + store AO as [bt][n][d] f16 (ready as GEMM A-matrix)
#pragma unroll
  for (int nt = 0; nt < 4; ++nt) {
    int col = h * HD + nt * 16 + lr;
#pragma unroll
    for (int r = 0; r < 8; ++r) {
      int qrow = q0 + r + half8;
      float val = o[nt][r] / lrow[r];
      AO[((size_t)bt * SEQ + qrow) * DMODEL + col] = (_Float16)val;
    }
  }
}

// ---------------------------------------------------------------------------
// Kernel 3: out = AO @ W_proj + b_proj (f32 out); tile 128x64, wave = 32x64
// ---------------------------------------------------------------------------
__global__ __launch_bounds__(128) void proj_kernel(
    const _Float16* __restrict__ A, const float* __restrict__ Wp,
    const float* __restrict__ bp, float* __restrict__ out) {
  __shared__ _Float16 As[128][32];
  __shared__ _Float16 Bs[64][32];
  const int t = threadIdx.x;
  const int w = t >> 5, lane = t & 31;
  const int m0 = blockIdx.y * 128;
  const int n0 = blockIdx.x * 64;
  const int lr = lane & 15;
  const int kbase = (lane < 16) ? 0 : 8;
  const int kb    = (lane < 16) ? 0 : 16;
  const int half8 = (lane < 16) ? 0 : 8;

  v8f acc0[4] = {};
  v8f acc1[4] = {};

  for (int k0 = 0; k0 < DMODEL; k0 += 32) {
    __syncthreads();
#pragma unroll
    for (int i = 0; i < 32; ++i) {
      int idx = i * 128 + t;
      int r = idx >> 5, c = idx & 31;
      As[r][c] = A[(size_t)(m0 + r) * DMODEL + (k0 + c)];
    }
#pragma unroll
    for (int i = 0; i < 16; ++i) {
      int idx = i * 128 + t;
      int nn = idx & 63, kk = idx >> 6;
      Bs[nn][kk] = (_Float16)Wp[(size_t)(k0 + kk) * DMODEL + (n0 + nn)];
    }
    __syncthreads();
    v16h a0 = load_a(&As[w * 32 + lr][0], kbase);
    v16h a1 = load_a(&As[w * 32 + 16 + lr][0], kbase);
#pragma unroll
    for (int nt = 0; nt < 4; ++nt) {
      v16h b = load_b(&Bs[nt * 16 + lr][0], kb);
      acc0[nt] = wmma16(a0, b, acc0[nt]);
      acc1[nt] = wmma16(a1, b, acc1[nt]);
    }
  }

#pragma unroll
  for (int nt = 0; nt < 4; ++nt) {
    int col = n0 + nt * 16 + lr;
    float bias = bp[col];
#pragma unroll
    for (int sub = 0; sub < 2; ++sub) {
      const v8f* acc = sub ? acc1 : acc0;
#pragma unroll
      for (int r = 0; r < 8; ++r) {
        int row = m0 + w * 32 + sub * 16 + r + half8;
        out[(size_t)row * DMODEL + col] = acc[nt][r] + bias;
      }
    }
  }
}

// ---------------------------------------------------------------------------
extern "C" void kernel_launch(void* const* d_in, const int* in_sizes, int n_in,
                              void* d_out, int out_size, void* d_ws,
                              size_t ws_size, hipStream_t stream) {
  const float* x  = (const float*)d_in[0];   // (2,8,1024,512)
  const float* Wq = (const float*)d_in[1];   // (512,1536)
  const float* bq = (const float*)d_in[2];   // (1536,)
  const float* Wp = (const float*)d_in[3];   // (512,512)
  const float* bp = (const float*)d_in[4];   // (512,)
  float* out = (float*)d_out;                // (2,8,1024,512) f32

  const size_t per = (size_t)BT * NH * SEQ * HD;  // 8M f16 elems each
  _Float16* Q  = (_Float16*)d_ws;
  _Float16* K  = Q + per;
  _Float16* V  = K + per;                         // stored transposed per head
  _Float16* AO = V + per;                         // [16384][512] f16
  // total workspace use: 4 * 8M * 2B = 64 MB

  qkv_kernel<<<dim3(D3 / 64, (BT * SEQ) / 128), 128, 0, stream>>>(x, Wq, bq,
                                                                  Q, K, V);
  attn_kernel<<<dim3(BT * NH * (SEQ / 64)), 128, 0, stream>>>(Q, K, V, AO);
  proj_kernel<<<dim3(DMODEL / 64, (BT * SEQ) / 128), 128, 0, stream>>>(AO, Wp,
                                                                       bp, out);
}
